// GlobalDensityApprox_18476949307782
// MI455X (gfx1250) — matrix operand order repaired and checked by reading, not compiled, not run on hardware
//
#include <hip/hip_runtime.h>

// ============================================================================
// GlobalDensityApprox forward for MI455X (gfx1250, wave32, WMMA).
//
// All GEMM-shaped work runs through one bf16 WMMA GEMM kernel
// (v_wmma_f32_16x16x32_bf16, f32 acc): 128x64 block tile, 8 waves, each wave
// owns 32x32 (4 WMMAs / K-step), double-buffered LDS, b128 global loads, and
// GLOBAL_LOAD_ASYNC_TO_LDS_B128 for the A-tile with s_wait_asynccnt.
// Split-K + f32 atomics for the [128xN]x[Nx128] attention reductions.
// CIDER O(N^2) kernel: 1 exp per pair for all 3 scales.
// ============================================================================

#define NTOK   4096   // N (tokens per batch)
#define DMODEL 128    // D
#define WMLP   256    // int(D * ENH)

typedef __attribute__((ext_vector_type(16))) __bf16 bf16x16;
typedef __attribute__((ext_vector_type(8)))  float  v8f;
typedef unsigned short u16;
typedef int v4i_gcc __attribute__((vector_size(16)));   // matches builtin's V4i param

#if defined(__HIP_DEVICE_COMPILE__) && \
    __has_builtin(__builtin_amdgcn_global_load_async_to_lds_b128) && \
    __has_builtin(__builtin_amdgcn_s_wait_asynccnt)
#define HAVE_ASYNC_LDS 1
#else
#define HAVE_ASYNC_LDS 0
#endif

__device__ __forceinline__ u16 f2bf(float f) {
  unsigned int x = __float_as_uint(f);
  unsigned int r = x + 0x7fffu + ((x >> 16) & 1u);   // round-to-nearest-even
  return (u16)(r >> 16);
}

// ---------------------------------------------------------------------------
// WMMA GEMM building blocks
// ---------------------------------------------------------------------------
// A-fragment (16x32 bf16) from LDS row-major tile (stride 40):
// lanes 0-15: M=lane, K {kb..kb+7, kb+16..kb+23} with kb=0; lanes 16-31: kb=8.
__device__ __forceinline__ bf16x16 frag_a_ld(const u16* sA, int mrow0, int lane) {
  const int row = mrow0 + (lane & 15);
  const int kb  = (lane < 16) ? 0 : 8;
  union { bf16x16 v; uint4 q[2]; } f;
  f.q[0] = *reinterpret_cast<const uint4*>(&sA[row * 40 + kb]);
  f.q[1] = *reinterpret_cast<const uint4*>(&sA[row * 40 + kb + 16]);
  return f.v;
}
// B-fragment (32x16 bf16) from LDS column-major (transposed) tile (stride 40):
// lane col = lane&15; lanes 0-15: K {0..7,16..23}, lanes 16-31: K {8..15,24..31}.
__device__ __forceinline__ bf16x16 frag_b_ld(const u16* sBt, int ncol0, int lane) {
  const int col = ncol0 + (lane & 15);
  const int kb  = (lane < 16) ? 0 : 8;
  union { bf16x16 v; uint4 q[2]; } f;
  f.q[0] = *reinterpret_cast<const uint4*>(&sBt[col * 40 + kb]);
  f.q[1] = *reinterpret_cast<const uint4*>(&sBt[col * 40 + kb + 16]);
  return f.v;
}

// Stage the 128x32 A tile into LDS. Async global->LDS when available
// (tracked by ASYNCcnt), else b128 load + ds_store fallback.
__device__ __forceinline__ void stage_A(const u16* __restrict__ Ab, u16* sA,
                                        int tileM, int K, int k0, int tid) {
#pragma unroll
  for (int t = 0; t < 2; ++t) {
    const int g  = tid + t * 256;          // 512 chunks of 8 u16
    const int ar = g >> 2, ak = (g & 3) * 8;
    const u16* src = &Ab[(size_t)(tileM + ar) * K + k0 + ak];
    u16* dst = &sA[ar * 40 + ak];
#if HAVE_ASYNC_LDS
    __builtin_amdgcn_global_load_async_to_lds_b128(
        (__attribute__((address_space(1))) v4i_gcc*)src,
        (__attribute__((address_space(3))) v4i_gcc*)dst, 0, 0);
#else
    if (k0 + 32 < K)
      __builtin_prefetch(src + 32, 0, 1);
    *reinterpret_cast<uint4*>(dst) = *reinterpret_cast<const uint4*>(src);
#endif
  }
}

__device__ __forceinline__ void wait_async_lds() {
#if HAVE_ASYNC_LDS
  __builtin_amdgcn_s_wait_asynccnt(0);
#endif
}

__device__ __forceinline__ uint4 fetch_B(const u16* __restrict__ Bb, int Nc,
                                         int tileN, int k0, int tid) {
  const int br = tid >> 3, bc = (tid & 7) * 8;   // 32 rows x 64 cols, 8 u16/thread
  return *reinterpret_cast<const uint4*>(&Bb[(size_t)(k0 + br) * Nc + tileN + bc]);
}
__device__ __forceinline__ void store_B(u16* sBt, uint4 v, int tid) {
  const int br = tid >> 3, bc = (tid & 7) * 8;
  union { uint4 q; u16 s[8]; } uw; uw.q = v;
#pragma unroll
  for (int u = 0; u < 8; ++u) sBt[(bc + u) * 40 + br] = uw.s[u];
}

// ---------------------------------------------------------------------------
// WMMA GEMM: C[M,N] = A[M,K](bf16) @ B[K,N](bf16) (+bias) (+residual)
// Block: 256 threads = 8 waves; tile 128x64, BK=32, double-buffered LDS.
// Wave (wm,wn), wm in 0..3, wn in 0..1: owns a 32x32 C slice -> 4 WMMA/K-step.
// splitK > 1: grid.z = batch*splitK, f32 atomicAdd epilogue (no bias/res/bf16).
// ---------------------------------------------------------------------------
__global__ void __launch_bounds__(256)
gemm_bf16_wmma(const u16* __restrict__ A, const u16* __restrict__ Bw,
               const float* __restrict__ bias, const float* __restrict__ residual,
               float* __restrict__ Cf, u16* __restrict__ Cb,
               int M, int Nc, int K,
               long long strideA, long long strideB, long long strideC,
               int splitK) {
  __shared__ u16 sA[2][128 * 40];
  __shared__ u16 sB[2][64 * 40];
  const int tid  = threadIdx.x;
  const int lane = tid & 31, wid = tid >> 5;
  const int wm = wid & 3, wn = wid >> 2;
  const int tileM = blockIdx.y * 128, tileN = blockIdx.x * 64;
  const int bz = (int)blockIdx.z / splitK;
  const int ks = (int)blockIdx.z % splitK;
  const int Klocal = K / splitK;
  const int kBase  = ks * Klocal;
  const int NK     = Klocal / 32;
  const u16* Ab = A + (size_t)bz * strideA;
  const u16* Bb = Bw + (size_t)bz * strideB;
  const size_t cbase = (size_t)bz * strideC;

  v8f acc[2][2] = {};

  // prologue: stage tile 0 into buffer 0
  stage_A(Ab, sA[0], tileM, K, kBase, tid);
  store_B(sB[0], fetch_B(Bb, Nc, tileN, kBase, tid), tid);
  wait_async_lds();
  __syncthreads();

  for (int kt = 0; kt < NK; ++kt) {
    const int cur = kt & 1, nxt = cur ^ 1;
    const bool hasNext = (kt + 1) < NK;
    uint4 breg{};
    if (hasNext) {
      const int k0 = kBase + (kt + 1) * 32;
      stage_A(Ab, sA[nxt], tileM, K, k0, tid);   // async (or ld/st) into other buffer
      breg = fetch_B(Bb, Nc, tileN, k0, tid);    // in flight during WMMAs
    }
    {
      const u16* cA = sA[cur];
      const u16* cB = sB[cur];
      bf16x16 a0 = frag_a_ld(cA, wm * 32, lane);
      bf16x16 a1 = frag_a_ld(cA, wm * 32 + 16, lane);
      bf16x16 b0 = frag_b_ld(cB, wn * 32, lane);
      bf16x16 b1 = frag_b_ld(cB, wn * 32 + 16, lane);
      acc[0][0] = __builtin_amdgcn_wmma_f32_16x16x32_bf16(false, a0, false, b0, (short)0, acc[0][0], false, false);
      acc[0][1] = __builtin_amdgcn_wmma_f32_16x16x32_bf16(false, a0, false, b1, (short)0, acc[0][1], false, false);
      acc[1][0] = __builtin_amdgcn_wmma_f32_16x16x32_bf16(false, a1, false, b0, (short)0, acc[1][0], false, false);
      acc[1][1] = __builtin_amdgcn_wmma_f32_16x16x32_bf16(false, a1, false, b1, (short)0, acc[1][1], false, false);
    }
    if (hasNext) store_B(sB[nxt], breg, tid);
    wait_async_lds();
    __syncthreads();
  }

  // C layout per 16x16 tile: lane 0-15 -> N=lane, M=r; lane 16-31 -> M=8+r.
  const int rsub = (lane < 16) ? 0 : 8;
  const int csub = lane & 15;
#pragma unroll
  for (int mi = 0; mi < 2; ++mi) {
#pragma unroll
    for (int ni = 0; ni < 2; ++ni) {
#pragma unroll
      for (int r = 0; r < 8; ++r) {
        const int row = tileM + wm * 32 + mi * 16 + rsub + r;
        const int col = tileN + wn * 32 + ni * 16 + csub;
        float v = acc[mi][ni][r];
        const size_t idx = cbase + (size_t)row * Nc + col;
        if (splitK > 1) {
          atomicAdd(&Cf[idx], v);
        } else {
          if (bias) v += bias[col];
          if (residual) v += residual[idx];
          if (Cf) Cf[idx] = v;
          if (Cb) Cb[idx] = f2bf(v);
        }
      }
    }
  }
}

// ---------------------------------------------------------------------------
// Small helper kernels
// ---------------------------------------------------------------------------
__device__ __forceinline__ float block_reduce256(float v, float* red) {
  const int t = threadIdx.x;
  red[t] = v; __syncthreads();
  for (int o = 128; o > 0; o >>= 1) { if (t < o) red[t] += red[t + o]; __syncthreads(); }
  float r = red[0]; __syncthreads();
  return r;
}

__global__ void k_f32_to_bf16(const float* __restrict__ src, u16* __restrict__ dst, int n) {
  int i = blockIdx.x * blockDim.x + threadIdx.x;
  if (i < n) dst[i] = f2bf(src[i]);
}

__global__ void k_wmean(const float* rho, const float* w, const float* coords,
                        float* wrho, float* means, int N) {
  const int b = blockIdx.x, t = threadIdx.x;
  __shared__ float red[256];
  float s0 = 0.f, sx = 0.f, sy = 0.f, sz = 0.f;
  for (int n = t; n < N; n += 256) {
    const float wr = w[b * N + n] * rho[b * N + n];
    wrho[b * N + n] = wr;
    const float* c = &coords[((size_t)b * N + n) * 3];
    s0 += wr; sx += wr * c[0]; sy += wr * c[1]; sz += wr * c[2];
  }
  s0 = block_reduce256(s0, red);
  sx = block_reduce256(sx, red);
  sy = block_reduce256(sy, red);
  sz = block_reduce256(sz, red);
  if (t == 0) {
    const float inv = 1.f / s0;
    means[b * 4 + 0] = sx * inv; means[b * 4 + 1] = sy * inv;
    means[b * 4 + 2] = sz * inv; means[b * 4 + 3] = s0;
  }
}

__global__ void k_wcov(const float* coords, const float* wrho, const float* means,
                       float* covs, int N) {
  const int b = blockIdx.x, t = threadIdx.x;
  __shared__ float red[256];
  const float mx = means[b*4], my = means[b*4+1], mz = means[b*4+2];
  const float inv = 1.f / means[b*4+3];
  float sxx=0, sxy=0, sxz=0, syy=0, syz=0, szz=0;
  for (int n = t; n < N; n += 256) {
    const float wr = wrho[b * N + n] * inv;
    const float* c = &coords[((size_t)b * N + n) * 3];
    const float x = c[0]-mx, y = c[1]-my, z = c[2]-mz;
    sxx += wr*x*x; sxy += wr*x*y; sxz += wr*x*z;
    syy += wr*y*y; syz += wr*y*z; szz += wr*z*z;
  }
  sxx = block_reduce256(sxx, red); sxy = block_reduce256(sxy, red);
  sxz = block_reduce256(sxz, red); syy = block_reduce256(syy, red);
  syz = block_reduce256(syz, red); szz = block_reduce256(szz, red);
  if (t == 0) {
    covs[b*9+0]=sxx; covs[b*9+1]=sxy; covs[b*9+2]=sxz;
    covs[b*9+3]=sxy; covs[b*9+4]=syy; covs[b*9+5]=syz;
    covs[b*9+6]=sxz; covs[b*9+7]=syz; covs[b*9+8]=szz;
  }
}

// Jacobi eigendecomposition of symmetric 3x3; eigenvectors as columns, sorted
// ascending. R stored [d][i] (column i = eigenvector i).
__global__ void k_eigh3(const float* covs, float* R, int Bc) {
  const int b = threadIdx.x;
  if (b >= Bc) return;
  float A[3][3], V[3][3] = {{1,0,0},{0,1,0},{0,0,1}};
  for (int i = 0; i < 3; ++i)
    for (int j = 0; j < 3; ++j) A[i][j] = covs[b*9 + i*3 + j];
  const int PQ[3][2] = {{0,1},{0,2},{1,2}};
  for (int sweep = 0; sweep < 10; ++sweep) {
    for (int r = 0; r < 3; ++r) {
      const int p = PQ[r][0], q = PQ[r][1];
      const float apq = A[p][q];
      if (fabsf(apq) < 1e-14f) continue;
      const float theta = (A[q][q] - A[p][p]) / (2.f * apq);
      const float tt = copysignf(1.f, theta) / (fabsf(theta) + sqrtf(theta*theta + 1.f));
      const float c = rsqrtf(tt*tt + 1.f), s = tt * c;
      for (int k = 0; k < 3; ++k) {
        const float akp = A[k][p], akq = A[k][q];
        A[k][p] = c*akp - s*akq; A[k][q] = s*akp + c*akq;
      }
      for (int k = 0; k < 3; ++k) {
        const float apk = A[p][k], aqk = A[q][k];
        A[p][k] = c*apk - s*aqk; A[q][k] = s*apk + c*aqk;
      }
      for (int k = 0; k < 3; ++k) {
        const float vkp = V[k][p], vkq = V[k][q];
        V[k][p] = c*vkp - s*vkq; V[k][q] = s*vkp + c*vkq;
      }
    }
  }
  int ord[3] = {0, 1, 2};
  for (int i = 0; i < 2; ++i)
    for (int j = 0; j < 2 - i; ++j)
      if (A[ord[j]][ord[j]] > A[ord[j+1]][ord[j+1]]) { int tmp = ord[j]; ord[j] = ord[j+1]; ord[j+1] = tmp; }
  for (int d = 0; d < 3; ++d)
    for (int i = 0; i < 3; ++i) R[b*9 + d*3 + i] = V[d][ord[i]];
}

// Rotate coords, compute s2 (reduced gradient) and CIDER exponent a.
__global__ void k_rotate(const float* coords, const float* means, const float* R,
                         const float* rho, const float* gamma,
                         float* coordsR, float* s2, float* a, int N, int total) {
  const int i = blockIdx.x * blockDim.x + threadIdx.x;
  if (i >= total) return;
  const int b = i / N;
  const float* c = &coords[(size_t)i * 3];
  const float cx = c[0]-means[b*4+0], cy = c[1]-means[b*4+1], cz = c[2]-means[b*4+2];
  const float* Rb = &R[b * 9];
  float* o = &coordsR[(size_t)i * 3];
  o[0] = cx*Rb[0] + cy*Rb[3] + cz*Rb[6];
  o[1] = cx*Rb[1] + cy*Rb[4] + cz*Rb[7];
  o[2] = cx*Rb[2] + cy*Rb[5] + cz*Rb[8];
  const float re = rho[i] + 1e-8f;
  const float t = cbrtf(re);
  const float r23 = t * t, r83 = re * re * r23;
  const float CG = 38.28312006f;                 // 4*(3*pi^2)^(2/3)
  const float s2v = gamma[i] / (CG * r83);
  s2[i] = s2v;
  a[i] = 4.f * r23 * (1.f + 4.f * s2v);          // A0=A1=4
}

// CIDER O(N^2) pair kernel; one exp per pair covers all three scales:
// exp(-s*a*d2) for s={0.25,1,4} = p, p^4, p^16 with p = exp(-0.25*a*d2).
__global__ void __launch_bounds__(256)
k_cider_partial(const float* __restrict__ coordsR, const float* __restrict__ wrho,
                const float* __restrict__ a,
                float* f025, float* f1, float* f4, int N) {
  const int b = blockIdx.y;
  const int i = blockIdx.x * 256 + threadIdx.x;
  const int jchunk = N / gridDim.z;
  const int jstart = blockIdx.z * jchunk;
  __shared__ float scx[256], scy[256], scz[256], sw[256];
  const size_t base = (size_t)b * N;
  const float ai = a[base + i];
  const float* ci = &coordsR[(base + i) * 3];
  const float cx = ci[0], cy = ci[1], cz = ci[2];
  float acc025 = 0.f, acc1 = 0.f, acc4 = 0.f;
  for (int j0 = jstart; j0 < jstart + jchunk; j0 += 256) {
    const int j = j0 + threadIdx.x;
    const float* cj = &coordsR[(base + j) * 3];
    scx[threadIdx.x] = cj[0]; scy[threadIdx.x] = cj[1]; scz[threadIdx.x] = cj[2];
    sw[threadIdx.x] = wrho[base + j];
    __syncthreads();
#pragma unroll 4
    for (int jj = 0; jj < 256; ++jj) {
      const float dx = cx - scx[jj], dy = cy - scy[jj], dz = cz - scz[jj];
      const float d2 = dx*dx + dy*dy + dz*dz;
      const float p  = __expf(-0.25f * ai * d2);
      const float p2 = p * p;
      const float e1 = p2 * p2;
      const float e2 = e1 * e1;
      const float e4 = e2 * e2;
      const float wj = sw[jj];
      acc025 += p * wj; acc1 += e1 * wj; acc4 += e4 * wj;
    }
    __syncthreads();
  }
  atomicAdd(&f025[base + i], acc025);
  atomicAdd(&f1[base + i], acc1);
  atomicAdd(&f4[base + i], acc4);
}

__global__ void k_cider_final(const float* s2, const float* a,
                              const float* f025, const float* f1, const float* f4,
                              float* x4, int total) {
  const int i = blockIdx.x * blockDim.x + threadIdx.x;
  if (i >= total) return;
  const float INVPI = 0.318309886f;
  const float ai = a[i];
  const float t0 = 0.25f * ai * INVPI, c0 = t0 * sqrtf(t0);
  const float t1 = ai * INVPI,         c1 = t1 * sqrtf(t1);
  const float t2 = 4.f * ai * INVPI,   c2 = t2 * sqrtf(t2);
  x4[(size_t)i*4+0] = logf(s2[i] + 1e-4f);
  x4[(size_t)i*4+1] = logf(c0 * f025[i] + 1e-4f);
  x4[(size_t)i*4+2] = logf(c1 * f1[i] + 1e-4f);
  x4[(size_t)i*4+3] = logf(c2 * f4[i] + 1e-4f);
}

// Embed layer 1 (K=4 -> VALU): h1 = tanh(x4 @ w1 + b1), bf16 out.
__global__ void k_embed1(const float* __restrict__ x4, const float* __restrict__ w1,
                         const float* __restrict__ b1, u16* __restrict__ h1, int total) {
  const int idx = blockIdx.x * blockDim.x + threadIdx.x;
  if (idx >= total) return;
  const int m = idx >> 8, o = idx & 255;
  const float* xr = &x4[(size_t)m * 4];
  float acc = b1[o];
#pragma unroll
  for (int d = 0; d < 4; ++d) acc += xr[d] * w1[d * 256 + o];
  h1[idx] = f2bf(tanhf(acc));
}

// LayerNorm (one wave per row of 128), bf16 out.
__global__ void k_ln_bf16(const float* __restrict__ x, const float* __restrict__ g,
                          const float* __restrict__ bta, u16* __restrict__ out) {
  const int row = blockIdx.x, lane = threadIdx.x;
  const float* xr = x + (size_t)row * DMODEL;
  float v[4]; float s = 0.f;
#pragma unroll
  for (int j = 0; j < 4; ++j) { v[j] = xr[lane + 32*j]; s += v[j]; }
#pragma unroll
  for (int o = 16; o > 0; o >>= 1) s += __shfl_xor(s, o, 32);
  const float mean = s * (1.f / DMODEL);
  float q = 0.f;
#pragma unroll
  for (int j = 0; j < 4; ++j) { const float d = v[j] - mean; q += d * d; }
#pragma unroll
  for (int o = 16; o > 0; o >>= 1) q += __shfl_xor(q, o, 32);
  const float rstd = rsqrtf(q * (1.f / DMODEL) + 1e-5f);
#pragma unroll
  for (int j = 0; j < 4; ++j) {
    const int c = lane + 32*j;
    out[(size_t)row * DMODEL + c] = f2bf((v[j] - mean) * rstd * g[c] + bta[c]);
  }
}

// Fourier-attention operand prep: phase=coords@omega^T; write qc/qs [n,f] and
// weighted kcT/ksT transposed [f,n] (A operand of the Sc/Ss reduction GEMMs).
__global__ void k_attn_prep(const float* __restrict__ coordsR, const float* __restrict__ omega,
                            const float* __restrict__ q, const float* __restrict__ k,
                            const float* __restrict__ w,
                            u16* qc, u16* qs, u16* kcT, u16* ksT, int N) {
  const int idx = blockIdx.x * blockDim.x + threadIdx.x;
  const int f = idx & 127;
  const int bn = idx >> 7;
  const int b = bn / N, n = bn - b * N;
  const float* cr = &coordsR[(size_t)bn * 3];
  const float* om = &omega[f * 3];
  const float ph = cr[0]*om[0] + cr[1]*om[1] + cr[2]*om[2];
  float sn, cs;
  __sincosf(ph, &sn, &cs);
  const size_t tok = (size_t)bn * DMODEL + f;
  const float qv = q[tok], kv = k[tok] * w[bn];
  qc[tok] = f2bf(qv * cs); qs[tok] = f2bf(qv * sn);
  const size_t tr = ((size_t)b * DMODEL + f) * N + n;
  kcT[tr] = f2bf(kv * cs); ksT[tr] = f2bf(kv * sn);
}

// Gated MLP nonlinearity: g = silu(u[:, :256]) * u[:, 256:], bf16 out.
__global__ void k_gate(const float* __restrict__ u, u16* __restrict__ g, int total) {
  const int idx = blockIdx.x * blockDim.x + threadIdx.x;
  if (idx >= total) return;
  const int m = idx >> 8, j = idx & 255;
  const float aa = u[(size_t)m * 512 + j];
  const float bb = u[(size_t)m * 512 + 256 + j];
  g[idx] = f2bf((aa / (1.f + __expf(-aa))) * bb);
}

// Final head: out[m] = sum_d tanh(h[m,d]) * wp[d] + bp (wave per row).
__global__ void k_final(const float* __restrict__ h, const float* __restrict__ wp,
                        const float* __restrict__ bp, float* __restrict__ out) {
  const int row = blockIdx.x, lane = threadIdx.x;
  float s = 0.f;
#pragma unroll
  for (int j = 0; j < 4; ++j) {
    const int d = lane + 32*j;
    s += tanhf(h[(size_t)row * DMODEL + d]) * wp[d];
  }
#pragma unroll
  for (int o = 16; o > 0; o >>= 1) s += __shfl_xor(s, o, 32);
  if (lane == 0) out[row] = s + bp[0];
}

// ============================================================================
// Host orchestration
// ============================================================================
extern "C" void kernel_launch(void* const* d_in, const int* in_sizes, int n_in,
                              void* d_out, int out_size, void* d_ws, size_t ws_size,
                              hipStream_t stream) {
  (void)n_in; (void)out_size; (void)ws_size;
  // Input order: rho, gamma, coords, weights, then params flattened as a JAX
  // pytree (dict keys sorted alphabetically, recursively):
  //   blocks[0..3]: attn{bk,bo,bq,bv,ln_b,ln_g,omega,wk,wo,wq,wv}, ln_b, ln_g,
  //                 mlp{b_in,b_out,w_in,w_out}
  //   embed{b1,b2,w1,w2}
  //   proj{b,ln_b,ln_g, mlp{b_in,b_out,w_in,w_out}, w}
  int ii = 0;
  const float* rho     = (const float*)d_in[ii++];
  const float* gamma   = (const float*)d_in[ii++];
  const float* coords  = (const float*)d_in[ii++];
  const float* weights = (const float*)d_in[ii++];
  struct BlkP {
    const float *bk,*bo,*bq,*bv,*aln_b,*aln_g,*omega,*wk,*wo,*wq,*wv;
    const float *ln_b,*ln_g, *b_in,*b_out,*w_in,*w_out;
  } bp[4];
  for (int b = 0; b < 4; ++b) {
    bp[b].bk    = (const float*)d_in[ii++];
    bp[b].bo    = (const float*)d_in[ii++];
    bp[b].bq    = (const float*)d_in[ii++];
    bp[b].bv    = (const float*)d_in[ii++];
    bp[b].aln_b = (const float*)d_in[ii++];
    bp[b].aln_g = (const float*)d_in[ii++];
    bp[b].omega = (const float*)d_in[ii++];
    bp[b].wk    = (const float*)d_in[ii++];
    bp[b].wo    = (const float*)d_in[ii++];
    bp[b].wq    = (const float*)d_in[ii++];
    bp[b].wv    = (const float*)d_in[ii++];
    bp[b].ln_b  = (const float*)d_in[ii++];
    bp[b].ln_g  = (const float*)d_in[ii++];
    bp[b].b_in  = (const float*)d_in[ii++];
    bp[b].b_out = (const float*)d_in[ii++];
    bp[b].w_in  = (const float*)d_in[ii++];
    bp[b].w_out = (const float*)d_in[ii++];
  }
  const float* e_b1   = (const float*)d_in[ii++];
  const float* e_b2   = (const float*)d_in[ii++];
  const float* e_w1   = (const float*)d_in[ii++];
  const float* e_w2   = (const float*)d_in[ii++];
  const float* p_b    = (const float*)d_in[ii++];
  const float* p_ln_b = (const float*)d_in[ii++];
  const float* p_ln_g = (const float*)d_in[ii++];
  const float* p_b_in = (const float*)d_in[ii++];
  const float* p_b_out= (const float*)d_in[ii++];
  const float* p_w_in = (const float*)d_in[ii++];
  const float* p_w_out= (const float*)d_in[ii++];
  const float* p_w    = (const float*)d_in[ii++];

  const int N  = NTOK;
  const int Bc = in_sizes[0] / N;     // batch (2)
  const int M  = Bc * N;              // total tokens (8192)
  if (Bc <= 0) return;

  char* wsp = (char*)d_ws;
  size_t off = 0;
  auto alloc = [&](size_t bytes) -> void* {
    void* p = wsp + off;
    off += (bytes + 255) & ~(size_t)255;
    return p;
  };

  float* wrho   = (float*)alloc((size_t)M * 4);
  float* means  = (float*)alloc((size_t)Bc * 4 * 4);
  float* covs   = (float*)alloc((size_t)Bc * 9 * 4);
  float* Rm     = (float*)alloc((size_t)Bc * 9 * 4);
  float* coordsR= (float*)alloc((size_t)M * 3 * 4);
  float* s2     = (float*)alloc((size_t)M * 4);
  float* aexp   = (float*)alloc((size_t)M * 4);
  float* fbuf   = (float*)alloc((size_t)3 * M * 4);
  float* f025 = fbuf, *f1 = fbuf + M, *f4 = fbuf + 2 * M;
  float* x4     = (float*)alloc((size_t)M * 4 * 4);
  u16*   h1b    = (u16*)  alloc((size_t)M * 256 * 2);
  float* phi    = (float*)alloc((size_t)M * DMODEL * 4);
  u16*   xnb    = (u16*)  alloc((size_t)M * DMODEL * 2);
  float* qf     = (float*)alloc((size_t)M * DMODEL * 4);
  float* kf     = (float*)alloc((size_t)M * DMODEL * 4);
  float* vf     = (float*)alloc((size_t)M * DMODEL * 4);
  u16*   vb     = (u16*)  alloc((size_t)M * DMODEL * 2);
  u16*   qc     = (u16*)  alloc((size_t)M * DMODEL * 2);
  u16*   qs     = (u16*)  alloc((size_t)M * DMODEL * 2);
  u16*   kcT    = (u16*)  alloc((size_t)M * DMODEL * 2);
  u16*   ksT    = (u16*)  alloc((size_t)M * DMODEL * 2);
  float* Sc     = (float*)alloc((size_t)Bc * DMODEL * DMODEL * 4);
  float* Ss     = (float*)alloc((size_t)Bc * DMODEL * DMODEL * 4);
  u16*   Scb    = (u16*)  alloc((size_t)Bc * DMODEL * DMODEL * 2);
  u16*   Ssb    = (u16*)  alloc((size_t)Bc * DMODEL * DMODEL * 2);
  float* attn   = (float*)alloc((size_t)M * DMODEL * 4);
  u16*   attnb  = (u16*)  alloc((size_t)M * DMODEL * 2);
  float* ubuf   = (float*)alloc((size_t)M * 512 * 4);
  u16*   gb     = (u16*)  alloc((size_t)M * 256 * 2);
  float* hbuf   = (float*)alloc((size_t)M * DMODEL * 4);
  // bf16 weight pool
  u16* w2b = (u16*)alloc(256 * 128 * 2);
  struct BlkB { u16 *wq, *wk, *wv, *wo, *win, *wout; } wb[4];
  for (int b = 0; b < 4; ++b) {
    wb[b].wq   = (u16*)alloc(128 * 128 * 2);
    wb[b].wk   = (u16*)alloc(128 * 128 * 2);
    wb[b].wv   = (u16*)alloc(128 * 128 * 2);
    wb[b].wo   = (u16*)alloc(128 * 128 * 2);
    wb[b].win  = (u16*)alloc(128 * 512 * 2);
    wb[b].wout = (u16*)alloc(256 * 128 * 2);
  }
  u16* pwin  = (u16*)alloc(128 * 512 * 2);
  u16* pwout = (u16*)alloc(256 * 128 * 2);

  auto conv = [&](const float* src, u16* dst, int n) {
    k_f32_to_bf16<<<(n + 255) / 256, 256, 0, stream>>>(src, dst, n);
  };
  auto gemm = [&](const u16* A, const u16* Bw, const float* bias, const float* resid,
                  float* Cf, u16* Cb, int Mm, int Nn, int Kk,
                  long long sA, long long sB, long long sC, int batch, int splitK) {
    dim3 grid(Nn / 64, Mm / 128, batch * splitK);
    gemm_bf16_wmma<<<grid, 256, 0, stream>>>(A, Bw, bias, resid, Cf, Cb, Mm, Nn, Kk,
                                             sA, sB, sC, splitK);
  };

  // Weight conversion to bf16 (every call; deterministic, graph-safe).
  conv(e_w2, w2b, 256 * 128);
  for (int b = 0; b < 4; ++b) {
    conv(bp[b].wq, wb[b].wq, 128 * 128);
    conv(bp[b].wk, wb[b].wk, 128 * 128);
    conv(bp[b].wv, wb[b].wv, 128 * 128);
    conv(bp[b].wo, wb[b].wo, 128 * 128);
    conv(bp[b].w_in,  wb[b].win,  128 * 512);
    conv(bp[b].w_out, wb[b].wout, 256 * 128);
  }
  conv(p_w_in, pwin, 128 * 512);
  conv(p_w_out, pwout, 256 * 128);

  // --- rotate_coords + embedding features -------------------------------
  k_wmean<<<Bc, 256, 0, stream>>>(rho, weights, coords, wrho, means, N);
  k_wcov <<<Bc, 256, 0, stream>>>(coords, wrho, means, covs, N);
  k_eigh3<<<1, 32, 0, stream>>>(covs, Rm, Bc);
  k_rotate<<<(M + 255) / 256, 256, 0, stream>>>(coords, means, Rm, rho, gamma,
                                                coordsR, s2, aexp, N, M);
  (void)hipMemsetAsync(fbuf, 0, (size_t)3 * M * sizeof(float), stream);
  {
    dim3 cg(N / 256, Bc, 4);
    k_cider_partial<<<cg, 256, 0, stream>>>(coordsR, wrho, aexp, f025, f1, f4, N);
  }
  k_cider_final<<<(M + 255) / 256, 256, 0, stream>>>(s2, aexp, f025, f1, f4, x4, M);
  k_embed1<<<(M * 256 + 255) / 256, 256, 0, stream>>>(x4, e_w1, e_b1, h1b, M * 256);
  gemm(h1b, w2b, e_b2, nullptr, phi, nullptr, M, 128, 256, 0, 0, 0, 1, 1);

  // --- transformer blocks -----------------------------------------------
  const long long sTok = (long long)N * DMODEL;       // per-batch token stride
  const long long sMat = (long long)DMODEL * DMODEL;  // per-batch S-matrix stride
  const long long sKT  = (long long)DMODEL * N;       // per-batch kT stride
  const size_t smatN = (size_t)Bc * DMODEL * DMODEL;
  for (int b = 0; b < 4; ++b) {
    k_ln_bf16<<<M, 32, 0, stream>>>(phi, bp[b].aln_g, bp[b].aln_b, xnb);
    gemm(xnb, wb[b].wq, bp[b].bq, nullptr, qf, nullptr, M, 128, 128, 0, 0, 0, 1, 1);
    gemm(xnb, wb[b].wk, bp[b].bk, nullptr, kf, nullptr, M, 128, 128, 0, 0, 0, 1, 1);
    gemm(xnb, wb[b].wv, bp[b].bv, nullptr, vf, vb,      M, 128, 128, 0, 0, 0, 1, 1);
    k_attn_prep<<<(M * DMODEL) / 256, 256, 0, stream>>>(coordsR, bp[b].omega, qf, kf,
                                                        weights, qc, qs, kcT, ksT, N);
    // Sc/Ss = (k*c*w)^T @ v : per-batch [128 x N] x [N x 128], split-K=8
    (void)hipMemsetAsync(Sc, 0, smatN * sizeof(float), stream);
    (void)hipMemsetAsync(Ss, 0, smatN * sizeof(float), stream);
    gemm(kcT, vb, nullptr, nullptr, Sc, nullptr, 128, 128, N, sKT, sTok, sMat, Bc, 8);
    gemm(ksT, vb, nullptr, nullptr, Ss, nullptr, 128, 128, N, sKT, sTok, sMat, Bc, 8);
    conv(Sc, Scb, (int)smatN);
    conv(Ss, Ssb, (int)smatN);
    // out = (q*c)@Sc + (q*s)@Ss : per-batch [N x 128] x [128 x 128]
    gemm(qc, Scb, nullptr, nullptr, attn, nullptr, N, 128, 128, sTok, sMat, sTok, Bc, 1);
    gemm(qs, Ssb, nullptr, attn,    attn, attnb,   N, 128, 128, sTok, sMat, sTok, Bc, 1);
    // phi += out @ wo + bo
    gemm(attnb, wb[b].wo, bp[b].bo, phi, phi, nullptr, M, 128, 128, 0, 0, 0, 1, 1);
    // phi += gated_mlp(LN(phi))
    k_ln_bf16<<<M, 32, 0, stream>>>(phi, bp[b].ln_g, bp[b].ln_b, xnb);
    gemm(xnb, wb[b].win, bp[b].b_in, nullptr, ubuf, nullptr, M, 512, 128, 0, 0, 0, 1, 1);
    k_gate<<<(M * 256 + 255) / 256, 256, 0, stream>>>(ubuf, gb, M * 256);
    gemm(gb, wb[b].wout, bp[b].b_out, phi, phi, nullptr, M, 128, 256, 0, 0, 0, 1, 1);
  }

  // --- projection head ---------------------------------------------------
  k_ln_bf16<<<M, 32, 0, stream>>>(phi, p_ln_g, p_ln_b, xnb);
  gemm(xnb, pwin, p_b_in, nullptr, ubuf, nullptr, M, 512, 128, 0, 0, 0, 1, 1);
  k_gate<<<(M * 256 + 255) / 256, 256, 0, stream>>>(ubuf, gb, M * 256);
  gemm(gb, pwout, p_b_out, nullptr, hbuf, nullptr, M, 128, 256, 0, 0, 0, 1, 1);
  k_final<<<M, 32, 0, stream>>>(hbuf, p_w, p_b, (float*)d_out);
}
